// PrefixSelfAttention_15650860826874
// MI455X (gfx1250) — matrix operand order repaired and checked by reading, not compiled, not run on hardware
//
#include <hip/hip_runtime.h>
#include <hip/hip_bf16.h>

typedef __attribute__((ext_vector_type(16))) __bf16 v16bf;
typedef __attribute__((ext_vector_type(8)))  float  v8f;

constexpr int   kNB    = 32;    // N*B
constexpr int   kH     = 16;
constexpr int   kD     = 64;
constexpr int   kT     = 16;
constexpr int   kS     = 1024;
constexpr int   kL     = 1024;
constexpr int   kE     = 1024;
constexpr int   kCols  = 2064;  // S + L + T
constexpr int   kPitch = 2080;  // padded LDS row pitch (floats)
constexpr float kScale = 0.125f;

__device__ __forceinline__ v8f wmma_bf16(v16bf a, v16bf b, v8f c) {
  return __builtin_amdgcn_wmma_f32_16x16x32_bf16(false, a, false, b, (short)0, c,
                                                 false, false);
}

// B fragment (32x16, K x N): lane holds col n=l&15, 16 contiguous K values
// starting at (l>>4)*16, f32 source converted to bf16.
__device__ __forceinline__ v16bf bfragB_from_f32(const float* __restrict__ p) {
  const float4* q = (const float4*)p;
  float4 x0 = q[0], x1 = q[1], x2 = q[2], x3 = q[3];
  v16bf r;
  r[0]  = (__bf16)x0.x; r[1]  = (__bf16)x0.y; r[2]  = (__bf16)x0.z; r[3]  = (__bf16)x0.w;
  r[4]  = (__bf16)x1.x; r[5]  = (__bf16)x1.y; r[6]  = (__bf16)x1.z; r[7]  = (__bf16)x1.w;
  r[8]  = (__bf16)x2.x; r[9]  = (__bf16)x2.y; r[10] = (__bf16)x2.z; r[11] = (__bf16)x2.w;
  r[12] = (__bf16)x3.x; r[13] = (__bf16)x3.y; r[14] = (__bf16)x3.z; r[15] = (__bf16)x3.w;
  return r;
}

// A fragment (16x32, M x K): lane holds row m=l&15; elements 0..15 = K kb8+0..15
// (kb8 = (l>>4)*16). Two contiguous runs of 8 f32.
__device__ __forceinline__ v16bf afragA_from_f32(const float* __restrict__ row, int kb8) {
  const float4* p0 = (const float4*)(row + kb8);
  const float4* p1 = (const float4*)(row + kb8 + 8);
  float4 x0 = p0[0], x1 = p0[1], y0 = p1[0], y1 = p1[1];
  v16bf r;
  r[0]  = (__bf16)x0.x; r[1]  = (__bf16)x0.y; r[2]  = (__bf16)x0.z; r[3]  = (__bf16)x0.w;
  r[4]  = (__bf16)x1.x; r[5]  = (__bf16)x1.y; r[6]  = (__bf16)x1.z; r[7]  = (__bf16)x1.w;
  r[8]  = (__bf16)y0.x; r[9]  = (__bf16)y0.y; r[10] = (__bf16)y0.z; r[11] = (__bf16)y0.w;
  r[12] = (__bf16)y1.x; r[13] = (__bf16)y1.y; r[14] = (__bf16)y1.z; r[15] = (__bf16)y1.w;
  return r;
}

// ---------------------------------------------------------------------------
// Kernel 1: QKV projection.  out = hs @ W^T + b, written as bf16 (nb,h,t,d).
// 2D-tiled: block = 4 waves (N-tile = 64 cols, waves share A rows via WGP$),
// each wave processes 4 nb values (M-tile = 64 rows) reusing its B fragment
// across 4 WMMAs per K-step.  grid = (E/64, 3, NB/4).
// ---------------------------------------------------------------------------
__global__ __launch_bounds__(128) void qkv_proj_kernel(
    const float* __restrict__ hs,
    const float* __restrict__ Wq, const float* __restrict__ bq,
    const float* __restrict__ Wk, const float* __restrict__ bk,
    const float* __restrict__ Wv, const float* __restrict__ bv,
    __bf16* __restrict__ q_ws, __bf16* __restrict__ k_ws, __bf16* __restrict__ v_ws)
{
  const int wave = threadIdx.x >> 5;
  const int lane = threadIdx.x & 31;
  const int jt   = blockIdx.x * 4 + wave;   // 16-wide output column tile
  const int mat  = blockIdx.y;              // 0=q, 1=k, 2=v
  const int nbg  = blockIdx.z * 4;          // first of 4 nb values

  const float* W    = (mat == 0) ? Wq : (mat == 1) ? Wk : Wv;
  const float* bias = (mat == 0) ? bq : (mat == 1) ? bk : bv;
  __bf16*      dst  = (mat == 0) ? q_ws : (mat == 1) ? k_ws : v_ws;

  const int n16  = lane & 15;
  const int half = lane >> 4;
  const int kb8  = half * 8;
  const int kb16 = half * 16;

  const int jn = jt * 16 + n16;                       // output column for B/C lanes
  const float* wrow   = W + (size_t)jn * kE;          // B col n = row jn of W
  const float* hsbase = hs + (size_t)nbg * kT * kE + (size_t)n16 * kE;

  v8f acc[4] = {v8f{}, v8f{}, v8f{}, v8f{}};
  for (int k0 = 0; k0 < kE; k0 += 32) {
    v16bf b = bfragB_from_f32(wrow + k0 + kb16);      // 1 B load, 4 WMMAs
#pragma unroll
    for (int i = 0; i < 4; ++i) {
      v16bf a = afragA_from_f32(hsbase + (size_t)i * kT * kE + k0, kb16);
      acc[i] = wmma_bf16(a, b, acc[i]);
    }
  }

  const float bv_ = bias[jn];
  const int h = jn >> 6;
  const int d = jn & 63;
#pragma unroll
  for (int i = 0; i < 4; ++i) {
    __bf16* drow = dst + (((size_t)(nbg + i) * kH + h) * kT) * kD + d;
#pragma unroll
    for (int r = 0; r < 8; ++r) {
      int t = r + half * 8;
      drow[(size_t)t * kD] = (__bf16)(acc[i][r] + bv_);
    }
  }
}

// ---------------------------------------------------------------------------
// Kernel 2: fused attention per (nb, h).  4 waves / block.
// LDS holds the full 16x2064 score matrix; softmax in place; context WMMA.
// ---------------------------------------------------------------------------
__global__ __launch_bounds__(128) void attn_kernel(
    const float* __restrict__ mask, const float* __restrict__ rel,
    const float* __restrict__ ppk,  const float* __restrict__ ppv,
    const float* __restrict__ pk,   const float* __restrict__ pv,
    const __bf16* __restrict__ q_ws, const __bf16* __restrict__ k_ws,
    const __bf16* __restrict__ v_ws, float* __restrict__ out)
{
  extern __shared__ float smem[];
  float* sc   = smem;                  // [16][kPitch] scores -> probs
  float* red  = smem + 16 * kPitch;    // [16][8] partial max
  float* red2 = red + 128;             // [16][8] partial sum

  const int nb   = blockIdx.x;
  const int h    = blockIdx.y;
  const int np   = nb >> 2;            // prefix batch index (nb / B)
  const int tid  = threadIdx.x;
  const int wave = tid >> 5;
  const int lane = tid & 31;
  const int n16  = lane & 15;
  const int half = lane >> 4;
  const int kb8  = half * 8;
  const int kb16 = half * 16;

  // -------- Q fragments (already bf16 in workspace) --------
  const __bf16* qrow = q_ws + (((size_t)nb * kH + h) * kT + n16) * kD;
  v16bf qa0, qa1;
#pragma unroll
  for (int e = 0; e < 16; ++e) {
    int k  = kb16 + e;
    qa0[e] = qrow[k];
    qa1[e] = qrow[32 + k];
  }

  const float* relbase = rel + (((size_t)nb * kH + h) * kT) * kCols;
  const float* maskrow = mask + (size_t)nb * kCols;

  // -------- scores: 129 tiles of 16 keys, waves stripe over tiles --------
  for (int jt = wave; jt < 129; jt += 4) {
    v8f acc = {};
    if (jt < 64) {                                    // prefix keys (f32)
      const float* kf = ppk + (((size_t)np * kH + h) * kS + (size_t)jt * 16 + n16) * kD;
      acc = wmma_bf16(qa0, bfragB_from_f32(kf + kb16), acc);
      acc = wmma_bf16(qa1, bfragB_from_f32(kf + 32 + kb16), acc);
    } else if (jt < 128) {                            // past keys (f32)
      const float* kf = pk + (((size_t)nb * kH + h) * kL + (size_t)(jt - 64) * 16 + n16) * kD;
      acc = wmma_bf16(qa0, bfragB_from_f32(kf + kb16), acc);
      acc = wmma_bf16(qa1, bfragB_from_f32(kf + 32 + kb16), acc);
    } else {                                          // new keys (bf16 ws)
      const __bf16* kb = k_ws + (((size_t)nb * kH + h) * kT + n16) * kD;
      v16bf b0, b1;
#pragma unroll
      for (int e = 0; e < 16; ++e) { b0[e] = kb[kb16 + e]; b1[e] = kb[32 + kb16 + e]; }
      acc = wmma_bf16(qa0, b0, acc);
      acc = wmma_bf16(qa1, b1, acc);
    }
    const int col  = jt * 16 + n16;
    const float mv = maskrow[col];
#pragma unroll
    for (int r = 0; r < 8; ++r) {
      int t = r + half * 8;
      sc[t * kPitch + col] = acc[r] * kScale + relbase[(size_t)t * kCols + col] + mv;
    }
  }
  __syncthreads();

  // -------- softmax over 2064 cols: 8 threads per row, 258 cols each -------
  {
    const int row   = tid & 15;
    const int chunk = tid >> 4;                       // 0..7, 8*258 == 2064
    float* srow = sc + row * kPitch + chunk * 258;
    float mx = -3.4e38f;
    for (int c = 0; c < 258; ++c) mx = fmaxf(mx, srow[c]);
    red[row * 8 + chunk] = mx;
    __syncthreads();
    float rmax = -3.4e38f;
#pragma unroll
    for (int i = 0; i < 8; ++i) rmax = fmaxf(rmax, red[row * 8 + i]);
    float s = 0.f;
    for (int c = 0; c < 258; ++c) {
      float e = __expf(srow[c] - rmax);
      srow[c] = e;
      s += e;
    }
    red2[row * 8 + chunk] = s;
    __syncthreads();
    float rsum = 0.f;
#pragma unroll
    for (int i = 0; i < 8; ++i) rsum += red2[row * 8 + i];
    const float inv = 1.0f / rsum;
    for (int c = 0; c < 258; ++c) srow[c] *= inv;
  }
  __syncthreads();

  // -------- context: each wave owns a 16-wide slice of D --------
  const int d0 = wave * 16;
  const int dn = d0 + n16;
  const float* prow = sc + n16 * kPitch;              // A row m = n16
  v8f acc = {};

  const float* vpre = ppv + (((size_t)np * kH + h) * kS) * kD;
  for (int ks = 0; ks < 32; ++ks) {                   // prefix V
    const int colbase = ks * 32;
    v16bf a, b;
#pragma unroll
    for (int e = 0; e < 16; ++e) {
      int k  = kb16 + e;
      a[e]   = (__bf16)prow[colbase + k];
      int kk = colbase + kb16 + e;
      b[e]   = (__bf16)vpre[(size_t)kk * kD + dn];
    }
    acc = wmma_bf16(a, b, acc);
  }
  const float* vpast = pv + (((size_t)nb * kH + h) * kL) * kD;
  for (int ks = 0; ks < 32; ++ks) {                   // past V
    const int colbase = kS + ks * 32;
    v16bf a, b;
#pragma unroll
    for (int e = 0; e < 16; ++e) {
      int k  = kb16 + e;
      a[e]   = (__bf16)prow[colbase + k];
      int kk = ks * 32 + kb16 + e;
      b[e]   = (__bf16)vpast[(size_t)kk * kD + dn];
    }
    acc = wmma_bf16(a, b, acc);
  }
  {                                                   // new V: only 16 valid K
    const int colbase = kS + kL;
    const __bf16* vb = v_ws + (((size_t)nb * kH + h) * kT) * kD;
    v16bf a, b;
#pragma unroll
    for (int e = 0; e < 16; ++e) {
      int k  = kb16 + e;
      a[e]   = (k < 16) ? (__bf16)prow[colbase + k] : (__bf16)0.f;
      int kk = kb16 + e;
      b[e]   = (kk < 16) ? vb[(size_t)kk * kD + dn] : (__bf16)0.f;
    }
    acc = wmma_bf16(a, b, acc);
  }

  // -------- store: out[nb][t][h*64 + d] (transpose folded into indexing) ----
#pragma unroll
  for (int r = 0; r < 8; ++r) {
    int t = r + half * 8;
    out[((size_t)nb * kT + t) * kE + h * kD + dn] = acc[r];
  }
}

// ---------------------------------------------------------------------------
extern "C" void kernel_launch(void* const* d_in, const int* in_sizes, int n_in,
                              void* d_out, int out_size, void* d_ws, size_t ws_size,
                              hipStream_t stream) {
  const float* hs   = (const float*)d_in[0];
  const float* mask = (const float*)d_in[1];
  const float* rel  = (const float*)d_in[2];
  const float* ppk  = (const float*)d_in[3];
  const float* ppv  = (const float*)d_in[4];
  const float* pk   = (const float*)d_in[5];
  const float* pv   = (const float*)d_in[6];
  const float* Wq   = (const float*)d_in[7];
  const float* bq   = (const float*)d_in[8];
  const float* Wk   = (const float*)d_in[9];
  const float* bk   = (const float*)d_in[10];
  const float* Wv   = (const float*)d_in[11];
  const float* bv   = (const float*)d_in[12];
  float* out = (float*)d_out;

  const size_t qkv_elems = (size_t)kNB * kH * kT * kD;   // 524288 bf16 = 1 MB each
  __bf16* q_ws = (__bf16*)d_ws;
  __bf16* k_ws = q_ws + qkv_elems;
  __bf16* v_ws = k_ws + qkv_elems;

  qkv_proj_kernel<<<dim3(kE / 64, 3, kNB / 4), 128, 0, stream>>>(
      hs, Wq, bq, Wk, bk, Wv, bv, q_ws, k_ws, v_ws);

  const size_t shmem = (size_t)(16 * kPitch) * sizeof(float) + 2 * 128 * sizeof(float);
  hipFuncSetAttribute(reinterpret_cast<const void*>(attn_kernel),
                      hipFuncAttributeMaxDynamicSharedMemorySize, (int)shmem);
  attn_kernel<<<dim3(kNB, kH), 128, shmem, stream>>>(
      mask, rel, ppk, ppv, pk, pv, q_ws, k_ws, v_ws, out);
}
